// SoftKMeansHead_18528488915429
// MI455X (gfx1250) — matrix operand order repaired and checked by reading, not compiled, not run on hardware
//
#include <hip/hip_runtime.h>

// ---------------------------------------------------------------------------
// SoftKMeansHead for MI455X (gfx1250, wave32, WMMA).
//
// Roofline: ~1.1 GFLOP of skinny GEMMs vs ~100 MB of unavoidable HBM traffic
// (64 MB input read once + 36 MB output) -> memory bound (~5 us at 23.3 TB/s).
// Strategy: single pass over the f32 input builds a 32 MB f16 transposed copy
// FT[b][n][c] (L2-resident on the 192 MB L2); all 4 softmax-assign passes and
// 4 proto passes then run v_wmma_f32_16x16x32_f16.  A-matrix staged in LDS
// (padded stride, conflict-free ds_load_b128, hoisted to VGPRs by the
// compiler), next tile's B row prefetched at WGP scope (inline
// global_prefetch_b8, scope 0 -> all cache levels) so the per-K-step
// s_wait_loadcnt hits the near cache; softmax done in the WMMA D-register
// layout with one shfl_xor(16) cross-lane step.
// ---------------------------------------------------------------------------

typedef __attribute__((ext_vector_type(16))) _Float16 v16h;
typedef __attribute__((ext_vector_type(8)))  _Float16 v8h;
typedef __attribute__((ext_vector_type(8)))  float    v8f;
typedef __attribute__((ext_vector_type(4)))  float    v4f;

#define B_   4
#define C_   256
#define H_   128
#define W_   128
#define N_   16384     // H_*W_
#define K16  16
#define OH_  512
#define OW_  512
#define TEMP_INV 1.25f // 1/0.8
#define EPS_ 1e-6f
#define ASTRIDE_H 264  // padded LDS row stride in halves (132 dwords) -> no bank conflicts

// WGP-scope prefetch (scope field omitted -> SCOPE 0: pull into all cache
// levels, unlike __builtin_prefetch which lowers to SCOPE_SE = GL2-only).
__device__ __forceinline__ void prefetch_wgp_512(const void* p)
{
    asm volatile("global_prefetch_b8 %0, off"            :: "v"(p));
    asm volatile("global_prefetch_b8 %0, off offset:128" :: "v"(p));
    asm volatile("global_prefetch_b8 %0, off offset:256" :: "v"(p));
    asm volatile("global_prefetch_b8 %0, off offset:384" :: "v"(p));
}

// ---------------------------------------------------------------------------
// k0: zero per-iteration accumulators/denominators, convert seed_w -> f16
// ---------------------------------------------------------------------------
__global__ void k0_init(float* __restrict__ protoacc, float* __restrict__ denoms,
                        const float* __restrict__ seedw, _Float16* __restrict__ seedwh)
{
    int i = blockIdx.x * 256 + threadIdx.x;
    if (i < 4 * B_ * K16 * C_) protoacc[i] = 0.f;   // 65536 floats
    if (i < 4 * B_ * K16)      denoms[i]   = 0.f;
    if (i < K16 * C_)          seedwh[i]   = (_Float16)seedw[i];
}

// ---------------------------------------------------------------------------
// k1: one HBM pass over input: build FT[b][n][c] (f16, transposed via LDS) and
//     f_sq[b][n] = sum_c f^2.  grid (N/64, B), block 256.
// ---------------------------------------------------------------------------
__global__ void k1_prep(const float* __restrict__ emb,
                        _Float16* __restrict__ FT, float* __restrict__ fsq)
{
    __shared__ _Float16 tile[C_ * 66];   // [c][n], row stride 66 halves (33 dw, odd -> conflict free)
    __shared__ float red[256];
    const int b  = blockIdx.y;
    const int n0 = blockIdx.x * 64;
    const int tid = threadIdx.x;
    const int nj = tid & 63, ci = tid >> 6;

    float acc = 0.f;
    for (int cc = 0; cc < C_; cc += 4) {
        int c = cc + ci;
        float v = emb[((size_t)(b * C_ + c)) * N_ + n0 + nj];   // coalesced along n
        acc += v * v;
        tile[c * 66 + nj] = (_Float16)v;
    }
    red[tid] = acc;
    __syncthreads();
    if (tid < 64)
        fsq[(size_t)b * N_ + n0 + tid] = red[tid] + red[tid + 64] + red[tid + 128] + red[tid + 192];

    // write FT[n][c] as packed u32 pairs, coalesced along c
    for (int it = 0; it < 32; ++it) {
        int flat = it * 256 + tid;
        int cp = flat & 127;         // c pair index
        int nl = flat >> 7;          // local n
        union { _Float16 h[2]; unsigned u; } pk;
        pk.h[0] = tile[(2 * cp)     * 66 + nl];
        pk.h[1] = tile[(2 * cp + 1) * 66 + nl];
        ((unsigned*)FT)[((size_t)(b * N_ + n0 + nl)) * (C_ / 2) + cp] = pk.u;
    }
}

// ---------------------------------------------------------------------------
// k2: logits GEMM (seeds or cross) + masked softmax over k in WMMA D layout.
//   D[kk][n] = sum_c A[kk][c] * F[c][n]  via v_wmma_f32_16x16x32_f16, K=C=256.
//   A (16x256 f16) staged in LDS once per block (hoisted to VGPRs); next
//   tile's per-lane B row prefetched at WGP scope during current tile's work.
//   mode 0: logits = D + bias   (seeds);  mode 1: logits = -(fsq+psq-2D)/T.
//   grid (64, B), block 128 (4 waves), 4 tiles of 16 columns per wave.
// ---------------------------------------------------------------------------
__global__ void __launch_bounds__(128, 1)
k2_assign(const _Float16* __restrict__ FT,
          const _Float16* __restrict__ seedwh,
          const _Float16* __restrict__ protoh,
          const float* __restrict__ fsq,
          const float* __restrict__ psq,
          const float* __restrict__ seedb,
          const int* __restrict__ kptr,
          _Float16* __restrict__ assignh,
          float* __restrict__ denom_it,
          float* __restrict__ outF,
          float* __restrict__ probsws,
          int mode, int isfinal)
{
    __shared__ unsigned AshU[K16 * (ASTRIDE_H / 2)];   // 16 rows x 132 dwords (264 halves)
    __shared__ float wden[K16];
    const int tid = threadIdx.x;
    const int b = blockIdx.y, g = blockIdx.x;
    const int kval = *kptr;

    // stage A into LDS (padded rows)
    {
        const _Float16* Ab = (mode == 0) ? seedwh : (protoh + (size_t)b * K16 * C_);
        const unsigned* src = (const unsigned*)Ab;
        for (int i = tid; i < K16 * (C_ / 2); i += 128) {
            int row = i >> 7, j = i & 127;
            AshU[row * (ASTRIDE_H / 2) + j] = src[i];
        }
    }
    if (tid < K16) wden[tid] = 0.f;
    __syncthreads();

    const int wave = tid >> 5, lane = tid & 31;
    const int col = lane & 15, hi = lane >> 4;

    // per-lane A pointer: row kk=col; K-halves {c0..c0+15} (+16 for hi)
    const _Float16* Ash = (const _Float16*)AshU;
    const v8h* ApL = (const v8h*)(Ash + col * ASTRIDE_H + hi * 16);  // ct -> [ct*4], [ct*4+1]

    float add8[8];
#pragma unroll
    for (int r = 0; r < 8; ++r)
        add8[r] = (mode == 0) ? seedb[hi * 8 + r] : psq[b * K16 + hi * 8 + r];

    float denacc[8] = {0, 0, 0, 0, 0, 0, 0, 0};
    size_t lbase = 0;
    if (isfinal) lbase = (size_t)B_ * kval * (OH_ * OW_) + (size_t)B_ * kval * C_;

    // prefetch first tile's per-lane B row into the near cache (WGP scope)
    {
        const int n0 = (g * 16 + wave * 4) * 16 + col;
        prefetch_wgp_512((const char*)(FT + ((size_t)b * N_ + n0) * C_));
    }

    for (int i = 0; i < 4; ++i) {
        const int nb = (g * 16 + wave * 4 + i) * 16;
        const int n = nb + col;

        // prefetch NEXT tile's per-lane B row (512 B) at WGP scope while this
        // tile runs its WMMA chain + softmax
        if (i < 3)
            prefetch_wgp_512((const char*)(FT + ((size_t)b * N_ + n + 16) * C_));

        const v8h* Bp = (const v8h*)(FT + ((size_t)b * N_ + n) * C_ + hi * 16);
        v8h cb0 = Bp[0], cb1 = Bp[1], nb0, nb1;
        v8f acc = {};
#pragma unroll
        for (int ct = 0; ct < 8; ++ct) {
            if (ct < 7) { nb0 = Bp[(ct + 1) * 4]; nb1 = Bp[(ct + 1) * 4 + 1]; }
            v8h a0 = ApL[ct * 4];
            v8h a1 = ApL[ct * 4 + 1];
            v16h Av, Bv;
#pragma unroll
            for (int j = 0; j < 8; ++j) {
                Av[j] = a0[j];   Av[8 + j] = a1[j];
                Bv[j] = cb0[j];  Bv[8 + j] = cb1[j];
            }
            acc = __builtin_amdgcn_wmma_f32_16x16x32_f16(false, Av, false, Bv,
                                                         (short)0, acc, false, false);
            cb0 = nb0; cb1 = nb1;
        }

        // D layout: VGPR r, lane -> row kk = r + 8*hi, col n.  Softmax over kk.
        const float fsqn = (mode == 1) ? fsq[(size_t)b * N_ + n] : 0.f;
        float lg[8], pr[8];
#pragma unroll
        for (int r = 0; r < 8; ++r) {
            float cr = acc[r];
            lg[r] = (mode == 0) ? (cr + add8[r])
                                : (-(fsqn + add8[r] - 2.f * cr) * TEMP_INV);
        }
        float m = -3.4e38f;
#pragma unroll
        for (int r = 0; r < 8; ++r) { int kk = r + 8 * hi; if (kk < kval) m = fmaxf(m, lg[r]); }
        m = fmaxf(m, __shfl_xor(m, 16));
        float s = 0.f;
#pragma unroll
        for (int r = 0; r < 8; ++r) {
            int kk = r + 8 * hi;
            float e = (kk < kval) ? __expf(lg[r] - m) : 0.f;
            pr[r] = e; s += e;
        }
        s += __shfl_xor(s, 16);
        const float inv = 1.f / s;
#pragma unroll
        for (int r = 0; r < 8; ++r) {
            pr[r] *= inv;
            int kk = r + 8 * hi;
            assignh[((size_t)b * K16 + kk) * N_ + n] = (_Float16)pr[r];
            float v = pr[r];                      // column-sum (over n) for denom
            v += __shfl_xor(v, 1); v += __shfl_xor(v, 2);
            v += __shfl_xor(v, 4); v += __shfl_xor(v, 8);
            denacc[r] += v;
        }
        if (isfinal) {
#pragma unroll
            for (int r = 0; r < 8; ++r) {
                int kk = r + 8 * hi;
                if (kk < kval) {
                    probsws[((size_t)b * K16 + kk) * N_ + n] = pr[r];
                    outF[lbase + ((size_t)b * kval + kk) * N_ + n] = lg[r];
                }
            }
        }
    }
    if (col == 0) {
#pragma unroll
        for (int r = 0; r < 8; ++r) atomicAdd(&wden[r + 8 * hi], denacc[r]);
    }
    __syncthreads();
    if (tid < K16) atomicAdd(&denom_it[b * K16 + tid], wden[tid]);
}

// ---------------------------------------------------------------------------
// k3: proto accumulation: D[kk][c] = sum_n assign[kk][n] * F[c][n] (K=N=16384).
//   A = assign_h rows (n-contiguous); B = raw f32 feats (n-contiguous per
//   lane), converted to f16 in registers.  32 WMMA K-steps per wave, loads
//   for step s+1 issued before the WMMA of step s; partial tiles combined
//   with global f32 atomics.  grid (16 ctiles, 16 chunks, B), block 32.
// ---------------------------------------------------------------------------
__global__ void __launch_bounds__(32, 1)
k3_proto(const float* __restrict__ emb,
         const _Float16* __restrict__ assignh,
         float* __restrict__ acc_it)
{
    const int ct16 = blockIdx.x, chunk = blockIdx.y, b = blockIdx.z;
    const int lane = threadIdx.x & 31;
    const int col = lane & 15, hi = lane >> 4;
    const int c = ct16 * 16 + col;

    const float*    Fb  = emb + ((size_t)b * C_ + c) * N_ + hi * 16;
    const _Float16* Ap0 = assignh + ((size_t)b * K16 + col) * N_ + hi * 16;
    const size_t nst = (size_t)chunk * 1024;

    // pipeline stage 0
    v8h a0 = *(const v8h*)(Ap0 + nst);
    v8h a1 = *(const v8h*)(Ap0 + nst + 8);
    v4f f0 = *(const v4f*)(Fb + nst);
    v4f f1 = *(const v4f*)(Fb + nst + 4);
    v4f f2 = *(const v4f*)(Fb + nst + 8);
    v4f f3 = *(const v4f*)(Fb + nst + 12);

    v8f acc = {};
    for (int s = 0; s < 32; ++s) {
        v8h na0, na1; v4f g0, g1, g2, g3;
        if (s < 31) {   // issue next K-step's loads before this step's WMMA
            const size_t n1 = nst + (size_t)(s + 1) * 32;
            na0 = *(const v8h*)(Ap0 + n1);
            na1 = *(const v8h*)(Ap0 + n1 + 8);
            g0 = *(const v4f*)(Fb + n1);
            g1 = *(const v4f*)(Fb + n1 + 4);
            g2 = *(const v4f*)(Fb + n1 + 8);
            g3 = *(const v4f*)(Fb + n1 + 12);
        }
        v16h Av, Bv;
#pragma unroll
        for (int j = 0; j < 8; ++j) { Av[j] = a0[j]; Av[8 + j] = a1[j]; }
#pragma unroll
        for (int j = 0; j < 4; ++j) {
            Bv[j]      = (_Float16)f0[j];  Bv[4 + j]  = (_Float16)f1[j];
            Bv[8 + j]  = (_Float16)f2[j];  Bv[12 + j] = (_Float16)f3[j];
        }
        acc = __builtin_amdgcn_wmma_f32_16x16x32_f16(false, Av, false, Bv,
                                                     (short)0, acc, false, false);
        a0 = na0; a1 = na1; f0 = g0; f1 = g1; f2 = g2; f3 = g3;
    }
#pragma unroll
    for (int r = 0; r < 8; ++r) {
        int kk = r + 8 * hi;
        atomicAdd(&acc_it[((size_t)b * K16 + kk) * C_ + c], acc[r]);
    }
}

// ---------------------------------------------------------------------------
// k4: proto = acc/(denom+eps); emit f16 proto for next cross pass, p_sq; on
//     final pass also write proto rows kk<k to d_out.  grid (16, B), block 256.
// ---------------------------------------------------------------------------
__global__ void k4_finalize(const float* __restrict__ acc_it,
                            const float* __restrict__ denom_it,
                            _Float16* __restrict__ protoh,
                            float* __restrict__ psq,
                            float* __restrict__ outF,
                            const int* __restrict__ kptr,
                            int isfinal)
{
    __shared__ float red[256];
    const int kk = blockIdx.x, b = blockIdx.y, c = threadIdx.x;
    const float d = denom_it[b * K16 + kk] + EPS_;
    const float p = acc_it[((size_t)b * K16 + kk) * C_ + c] / d;
    protoh[((size_t)b * K16 + kk) * C_ + c] = (_Float16)p;
    red[c] = p * p;
    __syncthreads();
    for (int off = 128; off > 0; off >>= 1) {
        if (c < off) red[c] += red[c + off];
        __syncthreads();
    }
    if (c == 0) psq[b * K16 + kk] = red[0];
    if (isfinal) {
        const int kval = *kptr;
        if (kk < kval) {
            const size_t pbase = (size_t)B_ * kval * (OH_ * OW_);
            outF[pbase + ((size_t)b * kval + kk) * C_ + c] = p;
        }
    }
}

// ---------------------------------------------------------------------------
// k5: x4 bilinear upsample, half-pixel centers with edge clamp
//     (matches jax.image.resize 'linear' edge renormalization for upsampling).
// ---------------------------------------------------------------------------
__global__ void k5_upsample(const float* __restrict__ probsws,
                            const int* __restrict__ kptr,
                            float* __restrict__ outF)
{
    const int kval = *kptr;
    const size_t flat = (size_t)blockIdx.x * 256 + threadIdx.x;
    const size_t total = (size_t)B_ * kval * OH_ * OW_;
    if (flat >= total) return;
    const int ox = (int)(flat & (OW_ - 1));
    const int oy = (int)((flat >> 9) & (OH_ - 1));
    const int t  = (int)(flat >> 18);
    const int kk = t % kval;
    const int b  = t / kval;

    const float fy = oy * 0.25f - 0.375f;
    const float fx = ox * 0.25f - 0.375f;
    const int iy0 = (int)floorf(fy), ix0 = (int)floorf(fx);
    const float wy = fy - (float)iy0, wx = fx - (float)ix0;
    const int y0 = max(iy0, 0), y1 = min(iy0 + 1, H_ - 1);
    const int x0 = max(ix0, 0), x1 = min(ix0 + 1, W_ - 1);

    const float* P = probsws + ((size_t)b * K16 + kk) * N_;
    const float v00 = P[y0 * W_ + x0], v01 = P[y0 * W_ + x1];
    const float v10 = P[y1 * W_ + x0], v11 = P[y1 * W_ + x1];
    outF[flat] = (1.f - wy) * ((1.f - wx) * v00 + wx * v01)
               +        wy  * ((1.f - wx) * v10 + wx * v11);
}

// ---------------------------------------------------------------------------
extern "C" void kernel_launch(void* const* d_in, const int* in_sizes, int n_in,
                              void* d_out, int out_size, void* d_ws, size_t ws_size,
                              hipStream_t stream)
{
    (void)in_sizes; (void)n_in; (void)out_size; (void)ws_size;
    const float* emb   = (const float*)d_in[0];
    const float* seedw = (const float*)d_in[1];
    const float* seedb = (const float*)d_in[2];
    const int*   kptr  = (const int*)d_in[3];
    float* outF = (float*)d_out;

    char* ws = (char*)d_ws;
    size_t off = 0;
    _Float16* FT      = (_Float16*)(ws + off); off += (size_t)B_ * C_ * N_ * 2;   // 32 MB
    float*    fsq     = (float*)(ws + off);    off += (size_t)B_ * N_ * 4;        // 256 KB
    _Float16* seedwh  = (_Float16*)(ws + off); off += (size_t)K16 * C_ * 2;       // 8 KB
    off = (off + 255) & ~(size_t)255;
    _Float16* assignh = (_Float16*)(ws + off); off += (size_t)B_ * K16 * N_ * 2;  // 2 MB
    float*    protoacc= (float*)(ws + off);    off += 4 * (size_t)B_ * K16 * C_ * 4; // 256 KB
    float*    denoms  = (float*)(ws + off);    off += 4 * (size_t)B_ * K16 * 4;
    off = (off + 255) & ~(size_t)255;
    _Float16* protoh  = (_Float16*)(ws + off); off += (size_t)B_ * K16 * C_ * 2;
    off = (off + 255) & ~(size_t)255;
    float*    psq     = (float*)(ws + off);    off += (size_t)B_ * K16 * 4;
    off = (off + 255) & ~(size_t)255;
    float*    probsws = (float*)(ws + off);    off += (size_t)B_ * K16 * N_ * 4;  // 4 MB

    k0_init<<<256, 256, 0, stream>>>(protoacc, denoms, seedw, seedwh);
    k1_prep<<<dim3(N_ / 64, B_), 256, 0, stream>>>(emb, FT, fsq);

    // pass 0: seeds -> assign0 -> proto1
    k2_assign<<<dim3(64, B_), 128, 0, stream>>>(FT, seedwh, protoh, fsq, psq, seedb, kptr,
                                                assignh, denoms, outF, probsws, 0, 0);
    k3_proto<<<dim3(16, 16, B_), 32, 0, stream>>>(emb, assignh, protoacc);
    k4_finalize<<<dim3(K16, B_), 256, 0, stream>>>(protoacc, denoms, protoh, psq, outF, kptr, 0);

    // 3 clustering iterations; last one emits logits_map + probs + proto out
    for (int it = 1; it <= 3; ++it) {
        const int fin = (it == 3);
        k2_assign<<<dim3(64, B_), 128, 0, stream>>>(FT, seedwh, protoh, fsq, psq, seedb, kptr,
                                                    assignh, denoms + (size_t)it * B_ * K16,
                                                    outF, probsws, 1, fin);
        k3_proto<<<dim3(16, 16, B_), 32, 0, stream>>>(emb, assignh,
                                                      protoacc + (size_t)it * B_ * K16 * C_);
        k4_finalize<<<dim3(K16, B_), 256, 0, stream>>>(protoacc + (size_t)it * B_ * K16 * C_,
                                                       denoms + (size_t)it * B_ * K16,
                                                       protoh, psq, outF, kptr, fin);
    }

    k5_upsample<<<(B_ * K16 * OH_ * OW_) / 256, 256, 0, stream>>>(probsws, kptr, outF);
}
